// Decoder_30485677867307
// MI455X (gfx1250) — matrix operand order repaired and checked by reading, not compiled, not run on hardware
//
#include <hip/hip_runtime.h>
#include <math.h>
#include <stdint.h>

typedef __bf16 bf16;
typedef __attribute__((ext_vector_type(16))) __bf16 bf16x16;
typedef __attribute__((ext_vector_type(8)))  float  f32x8;

#define B_SZ   512
#define SEQ    128
#define HID    256
#define EMBD   128
#define DMODEL 512
#define NHEAD  8
#define VOCAB  50000
#define EXTV   50400
#define ROWS   (B_SZ*SEQ)   // 65536

__device__ __forceinline__ bf16  f2bf(float x) { return (bf16)x; }
__device__ __forceinline__ float bf2f(bf16 x)  { return (float)x; }

// Async memory->LDS copy of 16 bytes (GLOBAL_LOAD_ASYNC_TO_LDS_B128, ASYNCcnt).
__device__ __forceinline__ void async_ld128(unsigned lds_off, const void* gaddr) {
    unsigned long long ga = (unsigned long long)(uintptr_t)gaddr;
    asm volatile("global_load_async_to_lds_b128 %0, %1, off"
                 :: "v"(lds_off), "v"(ga) : "memory");
}
__device__ __forceinline__ void wait_async0() {
    asm volatile("s_wait_asynccnt 0" ::: "memory");
}

// ---- WMMA fragment helpers (contiguous-k layouts, wave32) -------------------

// A-matrix 16x32 bf16: lane<16 -> row=lane, K k0..k0+15
//                      lane>=16 -> row=lane-16, K k0+16..k0+31
__device__ __forceinline__ bf16x16 load_a_frag(const bf16* base, int ld, int m0, int k0) {
    int lane = threadIdx.x & 31;
    const bf16* p = base + (size_t)(m0 + (lane & 15)) * ld + k0 + ((lane & 16) ? 16 : 0);
    bf16x16 f;
#pragma unroll
    for (int i = 0; i < 8; ++i) f[i] = p[i];
#pragma unroll
    for (int i = 0; i < 8; ++i) f[8 + i] = p[8 + i];
    return f;
}

// B-matrix 32x16 bf16 where B[k][n] = src[n][k] (src row-major, ld elems/row):
// lane<16 -> col=n0+lane, K k0..k0+15 ; lane>=16 -> col=n0+lane-16, K k0+16..k0+31
__device__ __forceinline__ bf16x16 load_b_frag(const bf16* base, int ld, int n0, int k0) {
    int lane = threadIdx.x & 31;
    const bf16* p = base + (size_t)(n0 + (lane & 15)) * ld + k0 + ((lane & 16) ? 16 : 0);
    bf16x16 f;
#pragma unroll
    for (int i = 0; i < 16; ++i) f[i] = p[i];
    return f;
}

// ---- small fp32 kernels ----------------------------------------------------

__global__ __launch_bounds__(256) void x_kernel(const int* __restrict__ y,
        const float* __restrict__ ct1, const float* __restrict__ embW,
        const float* __restrict__ xcW, const float* __restrict__ xcb,
        float* __restrict__ x) {
    int i = blockIdx.x * 256 + threadIdx.x;
    if (i >= B_SZ * EMBD) return;
    int b = i >> 7, e = i & 127;
    const float* w = xcW + (size_t)e * 640;
    const float* cr = ct1 + (size_t)b * DMODEL;
    float s = xcb[e];
    for (int k = 0; k < DMODEL; ++k) s += cr[k] * w[k];
    const float* er = embW + (size_t)y[b] * EMBD;
    for (int k = 0; k < EMBD; ++k) s += er[k] * w[DMODEL + k];
    x[i] = s;
}

__global__ __launch_bounds__(256) void gates_kernel(const float* __restrict__ x,
        const float* __restrict__ h0, const float* __restrict__ W_ih,
        const float* __restrict__ W_hh, const float* __restrict__ b_ih,
        const float* __restrict__ b_hh, float* __restrict__ gates) {
    int i = blockIdx.x * 256 + threadIdx.x;
    if (i >= B_SZ * 4 * HID) return;
    int b = i >> 10, j = i & 1023;
    float s = b_ih[j] + b_hh[j];
    const float* xr = x + (size_t)b * EMBD;
    const float* wi = W_ih + (size_t)j * EMBD;
    for (int k = 0; k < EMBD; ++k) s += xr[k] * wi[k];
    const float* hr = h0 + (size_t)b * HID;
    const float* wh = W_hh + (size_t)j * HID;
    for (int k = 0; k < HID; ++k) s += hr[k] * wh[k];
    gates[i] = s;
}

__global__ __launch_bounds__(256) void lstm_elem_kernel(const float* __restrict__ gates,
        const float* __restrict__ c0, float* __restrict__ h_new, float* __restrict__ c_new) {
    int i = blockIdx.x * 256 + threadIdx.x;
    if (i >= B_SZ * HID) return;
    int b = i >> 8, j = i & 255;
    const float* g = gates + (size_t)b * 1024;
    float ig = 1.f / (1.f + expf(-g[j]));
    float fg = 1.f / (1.f + expf(-g[HID + j]));
    float gg = tanhf(g[2 * HID + j]);
    float og = 1.f / (1.f + expf(-g[3 * HID + j]));
    float c  = fg * c0[i] + ig * gg;
    c_new[i] = c;
    h_new[i] = og * tanhf(c);
}

__global__ __launch_bounds__(256) void ct_kernel(const float* __restrict__ ctxmean,
        const float* __restrict__ Wo, float* __restrict__ c_t) {
    int i = blockIdx.x * 256 + threadIdx.x;
    if (i >= B_SZ * DMODEL) return;
    int b = i >> 9, j = i & 511;
    const float* a = ctxmean + (size_t)b * DMODEL;
    const float* w = Wo + (size_t)j * DMODEL;
    float s = 0.f;
    for (int k = 0; k < DMODEL; ++k) s += a[k] * w[k];
    c_t[i] = s;
}

__global__ __launch_bounds__(256) void o1_kernel(const float* __restrict__ h_new,
        const float* __restrict__ c_t, const float* __restrict__ o1W,
        const float* __restrict__ o1b, float* __restrict__ out1) {
    int i = blockIdx.x * 256 + threadIdx.x;
    if (i >= B_SZ * HID) return;
    int b = i >> 8, j = i & 255;
    const float* w = o1W + (size_t)j * 768;
    float s = o1b[j];
    const float* hr = h_new + (size_t)b * HID;
    for (int k = 0; k < HID; ++k) s += hr[k] * w[k];
    const float* cr = c_t + (size_t)b * DMODEL;
    for (int k = 0; k < DMODEL; ++k) s += cr[k] * w[HID + k];
    out1[i] = s;
}

__global__ __launch_bounds__(256) void pgen_kernel(const float* __restrict__ c_t,
        const float* __restrict__ h_new, const float* __restrict__ c_new,
        const float* __restrict__ x, const float* __restrict__ pgW,
        const float* __restrict__ pgb, float* __restrict__ p_gen) {
    int b = blockIdx.x * 256 + threadIdx.x;
    if (b >= B_SZ) return;
    float s = pgb[0];
    const float* cr = c_t + (size_t)b * DMODEL;
    for (int k = 0; k < DMODEL; ++k) s += cr[k] * pgW[k];
    const float* hr = h_new + (size_t)b * HID;
    for (int k = 0; k < HID; ++k) s += hr[k] * pgW[DMODEL + k];
    const float* cn = c_new + (size_t)b * HID;
    for (int k = 0; k < HID; ++k) s += cn[k] * pgW[DMODEL + HID + k];
    const float* xr = x + (size_t)b * EMBD;
    for (int k = 0; k < EMBD; ++k) s += xr[k] * pgW[DMODEL + 2 * HID + k];
    p_gen[b] = 1.f / (1.f + expf(-s));
}

// ---- one-time fp32 -> bf16 conversion of Wq/Wk/Wv --------------------------

__global__ __launch_bounds__(256) void convert_w_kernel(const float* __restrict__ Wq,
        const float* __restrict__ Wk, const float* __restrict__ Wv,
        bf16* __restrict__ Wqb, bf16* __restrict__ Wkb, bf16* __restrict__ Wvb) {
    int i = blockIdx.x * 256 + threadIdx.x;
    const int L = DMODEL * DMODEL;   // 262144
    if (i < L)           Wqb[i]         = f2bf(Wq[i]);
    else if (i < 2 * L)  Wkb[i - L]     = f2bf(Wk[i - L]);
    else if (i < 3 * L)  Wvb[i - 2 * L] = f2bf(Wv[i - 2 * L]);
}

// ---- WMMA GEMM: fused Q/K/V projection (out = enc @ W.T, stored bf16) ------
// A (enc fp32) converted via VALU staging; B (bf16 weights) async-loaded to
// LDS; C tile staged through LDS for coalesced b128 stores.

__global__ __launch_bounds__(256) void qkv_proj_kernel(const float* __restrict__ enc,
        const bf16* __restrict__ Wqb, const bf16* __restrict__ Wkb,
        const bf16* __restrict__ Wvb, bf16* __restrict__ Qb, bf16* __restrict__ Kb,
        bf16* __restrict__ Vb) {
    extern __shared__ char smem[];
    bf16* As = (bf16*)smem;            // 64*64*2 = 8192
    bf16* Bs = (bf16*)(smem + 8192);   // 8192
    const unsigned lds0 = __builtin_amdgcn_groupstaticsize();
    const unsigned bs_off = lds0 + 8192;

    const bf16* W = (blockIdx.z == 0) ? Wqb : (blockIdx.z == 1) ? Wkb : Wvb;
    bf16* Out     = (blockIdx.z == 0) ? Qb : (blockIdx.z == 1) ? Kb : Vb;
    int m0 = blockIdx.x * 64;
    int n0 = blockIdx.y * 64;
    int tid = threadIdx.x;
    int wave = tid >> 5;
    int t0 = 2 * wave, t1 = t0 + 1;
    int tm = (t0 >> 2) * 16;
    int tn0 = (t0 & 3) * 16, tn1 = (t1 & 3) * 16;
    f32x8 acc0 = {}, acc1 = {};
    for (int kc = 0; kc < DMODEL; kc += 64) {
        // B tile: async bf16 copy, 512 x 16B chunks
        for (int i = tid; i < 512; i += 256) {
            int r = i >> 3, c = i & 7;
            async_ld128(bs_off + i * 16, W + (size_t)(n0 + r) * DMODEL + kc + c * 8);
        }
        // A tile: fp32 load + convert (prefetch next chunk)
        if (kc + 64 < DMODEL)
            __builtin_prefetch(&enc[(size_t)(m0 + (tid >> 2)) * DMODEL + kc + 64], 0, 1);
        for (int i = tid; i < 64 * 64; i += 256) {
            int r = i >> 6, c = i & 63;
            As[i] = f2bf(enc[(size_t)(m0 + r) * DMODEL + kc + c]);
        }
        wait_async0();
        __syncthreads();
#pragma unroll
        for (int kk = 0; kk < 64; kk += 32) {
            bf16x16 a  = load_a_frag(As, 64, tm, kk);
            bf16x16 b0 = load_b_frag(Bs, 64, tn0, kk);
            bf16x16 b1 = load_b_frag(Bs, 64, tn1, kk);
            acc0 = __builtin_amdgcn_wmma_f32_16x16x32_bf16(false, a, false, b0, (short)0, acc0, false, false);
            acc1 = __builtin_amdgcn_wmma_f32_16x16x32_bf16(false, a, false, b1, (short)0, acc1, false, false);
        }
        __syncthreads();
    }
    // stage C tile in LDS (reuse As) for coalesced b128 stores
    bf16* Cs = As;
    int lane = tid & 31;
    int nn = lane & 15, mb = (lane >> 4) * 8;
#pragma unroll
    for (int r = 0; r < 8; ++r) {
        Cs[(tm + mb + r) * 64 + tn0 + nn] = f2bf(acc0[r]);
        Cs[(tm + mb + r) * 64 + tn1 + nn] = f2bf(acc1[r]);
    }
    __syncthreads();
    for (int i = tid; i < 512; i += 256) {
        int r = i >> 3, c = i & 7;
        *(uint4*)(&Out[(size_t)(m0 + r) * DMODEL + n0 + c * 8]) =
            *(const uint4*)(&Cs[r * 64 + c * 8]);
    }
}

// ---- attention: async-staged Q/K/V, WMMA scores, softmax, column-mean ------
// one block per (b, h); mean over s commutes through attn@v and Wo.

__global__ __launch_bounds__(256) void attn_kernel(const bf16* __restrict__ Qb,
        const bf16* __restrict__ Kb, const bf16* __restrict__ Vb,
        float* __restrict__ ctxmean) {
    extern __shared__ char smem[];
    float* sc = (float*)smem;                 // 128*128*4 = 65536
    bf16*  qs = (bf16*)(smem + 65536);        // 128*64*2  = 16384
    bf16*  ks = (bf16*)(smem + 81920);        // 16384
    bf16*  vs = (bf16*)(smem + 98304);        // 16384
    float* cm = (float*)(smem + 114688);      // 512
    const unsigned lds0 = __builtin_amdgcn_groupstaticsize();
    const unsigned qs_off = lds0 + 65536;
    const unsigned ks_off = lds0 + 81920;
    const unsigned vs_off = lds0 + 98304;

    int b = blockIdx.x, h = blockIdx.y;
    int tid = threadIdx.x;
    const bf16* gq = Qb + (size_t)b * SEQ * DMODEL + h * 64;
    const bf16* gk = Kb + (size_t)b * SEQ * DMODEL + h * 64;
    const bf16* gv = Vb + (size_t)b * SEQ * DMODEL + h * 64;
    // 1024 x 16B chunks per matrix, async memory->LDS
    for (int i = tid; i < 1024; i += 256) {
        int s = i >> 3, c = i & 7;
        size_t goff = (size_t)s * DMODEL + c * 8;
        unsigned l = (unsigned)i * 16;
        async_ld128(qs_off + l, gq + goff);
        async_ld128(ks_off + l, gk + goff);
        async_ld128(vs_off + l, gv + goff);
    }
    wait_async0();
    __syncthreads();

    int wave = tid >> 5;
    int tm = wave * 16;
    f32x8 acc[8] = {};
#pragma unroll
    for (int kk = 0; kk < 64; kk += 32) {
        bf16x16 a = load_a_frag(qs, 64, tm, kk);
#pragma unroll
        for (int j = 0; j < 8; ++j) {
            bf16x16 bb = load_b_frag(ks, 64, j * 16, kk);
            acc[j] = __builtin_amdgcn_wmma_f32_16x16x32_bf16(false, a, false, bb, (short)0, acc[j], false, false);
        }
    }
    int lane = tid & 31;
    int nn = lane & 15, mb = (lane >> 4) * 8;
    const float scale = 0.125f;   // 1/sqrt(64)
#pragma unroll
    for (int j = 0; j < 8; ++j)
#pragma unroll
        for (int r = 0; r < 8; ++r)
            sc[(tm + mb + r) * SEQ + j * 16 + nn] = acc[j][r] * scale;
    __syncthreads();
    if (tid < SEQ) {
        float* row = sc + tid * SEQ;
        float m = row[0];
        for (int t = 1; t < SEQ; ++t) m = fmaxf(m, row[t]);
        float ssum = 0.f;
        for (int t = 0; t < SEQ; ++t) { float e = expf(row[t] - m); row[t] = e; ssum += e; }
        float inv = 1.f / ssum;
        for (int t = 0; t < SEQ; ++t) row[t] *= inv;
    }
    __syncthreads();
    if (tid < SEQ) {
        float s = 0.f;
        for (int r = 0; r < SEQ; ++r) s += sc[r * SEQ + tid];
        cm[tid] = s * (1.0f / SEQ);
    }
    __syncthreads();
    if (tid < 64) {
        float s = 0.f;
        for (int t = 0; t < SEQ; ++t)
            s += cm[t] * bf2f(vs[t * 64 + tid]);
        ctxmean[(size_t)b * DMODEL + h * 64 + tid] = s;
    }
}

// ---- WMMA GEMM: logits = out1 @ o2_W.T + o2_b ------------------------------

__global__ __launch_bounds__(256) void logits_kernel(const float* __restrict__ out1,
        const float* __restrict__ o2W, const float* __restrict__ o2b,
        float* __restrict__ logits) {
    int m0 = blockIdx.x * 64;
    int n0 = blockIdx.y * 64;
    __shared__ bf16 As[64 * 256];   // 32KB
    __shared__ bf16 Bs[64 * 256];   // 32KB
    int tid = threadIdx.x;
    for (int i = tid; i < 64 * 256; i += 256) {
        int r = i >> 8, c = i & 255;
        As[i] = f2bf(out1[(size_t)(m0 + r) * HID + c]);
        int n = n0 + r;
        Bs[i] = (n < VOCAB) ? f2bf(o2W[(size_t)n * HID + c]) : f2bf(0.0f);
    }
    __syncthreads();
    int wave = tid >> 5;
    int t0 = 2 * wave, t1 = t0 + 1;
    int tm = (t0 >> 2) * 16;
    int tn0 = (t0 & 3) * 16, tn1 = (t1 & 3) * 16;
    f32x8 acc0 = {}, acc1 = {};
#pragma unroll
    for (int kk = 0; kk < HID; kk += 32) {
        bf16x16 a  = load_a_frag(As, 256, tm, kk);
        bf16x16 b0 = load_b_frag(Bs, 256, tn0, kk);
        bf16x16 b1 = load_b_frag(Bs, 256, tn1, kk);
        acc0 = __builtin_amdgcn_wmma_f32_16x16x32_bf16(false, a, false, b0, (short)0, acc0, false, false);
        acc1 = __builtin_amdgcn_wmma_f32_16x16x32_bf16(false, a, false, b1, (short)0, acc1, false, false);
    }
    int lane = tid & 31;
    int nn = lane & 15, mb = (lane >> 4) * 8;
#pragma unroll
    for (int r = 0; r < 8; ++r) {
        int m = m0 + tm + mb + r;
        int nA = n0 + tn0 + nn, nB = n0 + tn1 + nn;
        if (nA < VOCAB) logits[(size_t)m * VOCAB + nA] = acc0[r] + o2b[nA];
        if (nB < VOCAB) logits[(size_t)m * VOCAB + nB] = acc1[r] + o2b[nB];
    }
}

// ---- final: softmax over vocab, pointer-gen merge, scatter-add -------------

__global__ __launch_bounds__(256) void final_kernel(const float* __restrict__ logits,
        const float* __restrict__ p_gen, const float* __restrict__ c_t,
        const int* __restrict__ ext_idx, float* __restrict__ out) {
    int b = blockIdx.x;
    __shared__ float red[256];
    const float* row = logits + (size_t)b * VOCAB;
    float m = -INFINITY;
    for (int v = threadIdx.x; v < VOCAB; v += 256) m = fmaxf(m, row[v]);
    red[threadIdx.x] = m;
    __syncthreads();
    for (int s = 128; s > 0; s >>= 1) {
        if (threadIdx.x < s) red[threadIdx.x] = fmaxf(red[threadIdx.x], red[threadIdx.x + s]);
        __syncthreads();
    }
    m = red[0];
    __syncthreads();
    float ssum = 0.f;
    for (int v = threadIdx.x; v < VOCAB; v += 256) ssum += expf(row[v] - m);
    red[threadIdx.x] = ssum;
    __syncthreads();
    for (int s = 128; s > 0; s >>= 1) {
        if (threadIdx.x < s) red[threadIdx.x] += red[threadIdx.x + s];
        __syncthreads();
    }
    float inv = 1.f / red[0];
    float pg = p_gen[b];
    float* orow = out + (size_t)b * EXTV;
    for (int v = threadIdx.x; v < VOCAB; v += 256) orow[v] = pg * expf(row[v] - m) * inv;
    for (int v = VOCAB + threadIdx.x; v < EXTV; v += 256) orow[v] = 0.f;
    __syncthreads();
    float w = 1.f - pg;
    for (int j = threadIdx.x; j < DMODEL; j += 256)
        atomicAdd(&orow[ext_idx[(size_t)b * DMODEL + j]], w * c_t[(size_t)b * DMODEL + j]);
}

// ---- remaining tuple outputs ----------------------------------------------

__global__ __launch_bounds__(256) void aux_out_kernel(const float* __restrict__ h_new,
        const float* __restrict__ c_new, const float* __restrict__ c_t,
        const float* __restrict__ p_gen, const float* __restrict__ coverage,
        float* __restrict__ out) {
    int i = blockIdx.x * 256 + threadIdx.x;
    const int Lh = B_SZ * HID;      // 131072
    const int Lct = B_SZ * DMODEL;  // 262144
    const int Lpg = B_SZ;           // 512
    const int Lcov = B_SZ * SEQ;    // 65536
    float v;
    if (i < Lh)                       v = h_new[i];
    else if (i < 2 * Lh)              v = c_new[i - Lh];
    else if (i < 2 * Lh + Lct)        v = c_t[i - 2 * Lh];
    else if (i < 2 * Lh + 2 * Lct)    v = c_t[i - 2 * Lh - Lct];
    else if (i < 2 * Lh + 2 * Lct + Lpg)        v = p_gen[i - 2 * Lh - 2 * Lct];
    else if (i < 2 * Lh + 2 * Lct + Lpg + Lcov) v = coverage[i - 2 * Lh - 2 * Lct - Lpg];
    else return;
    out[25804800 + (size_t)i] = v;
}

extern "C" void kernel_launch(void* const* d_in, const int* in_sizes, int n_in,
                              void* d_out, int out_size, void* d_ws, size_t ws_size,
                              hipStream_t stream) {
    (void)in_sizes; (void)n_in; (void)out_size; (void)ws_size;
    const int*   y_t_1    = (const int*)d_in[0];
    const float* h0       = (const float*)d_in[1];
    const float* c0       = (const float*)d_in[2];
    const float* enc      = (const float*)d_in[3];
    const float* c_t_1    = (const float*)d_in[6];
    const int*   extvoc   = (const int*)d_in[8];
    const float* coverage = (const float*)d_in[9];
    const float* emb_W    = (const float*)d_in[11];
    const float* xc_W     = (const float*)d_in[12];
    const float* xc_b     = (const float*)d_in[13];
    const float* W_ih     = (const float*)d_in[14];
    const float* W_hh     = (const float*)d_in[15];
    const float* b_ih     = (const float*)d_in[16];
    const float* b_hh     = (const float*)d_in[17];
    const float* Wq       = (const float*)d_in[18];
    const float* Wk       = (const float*)d_in[19];
    const float* Wv       = (const float*)d_in[20];
    const float* Wo       = (const float*)d_in[21];
    const float* pg_W     = (const float*)d_in[22];
    const float* pg_b     = (const float*)d_in[23];
    const float* o1_W     = (const float*)d_in[24];
    const float* o1_b     = (const float*)d_in[25];
    const float* o2_W     = (const float*)d_in[26];
    const float* o2_b     = (const float*)d_in[27];

    char* ws = (char*)d_ws;
    // QKV bf16 (dead after attn) aliased later by fp32 logits
    bf16*  Qb      = (bf16*)(ws);
    bf16*  Kb      = (bf16*)(ws + 67108864ull);
    bf16*  Vb      = (bf16*)(ws + 134217728ull);
    float* logits  = (float*)(ws);                 // 512*50000*4 = 102.4 MB, alias
    const size_t base = 201326592ull;
    float* ctxmean = (float*)(ws + base);
    float* c_t     = (float*)(ws + base + 1048576ull);
    float* x       = (float*)(ws + base + 2097152ull);
    float* gates   = (float*)(ws + base + 2359296ull);
    float* h_new   = (float*)(ws + base + 4456448ull);
    float* c_new   = (float*)(ws + base + 4980736ull);
    float* out1    = (float*)(ws + base + 5505024ull);
    float* p_gen   = (float*)(ws + base + 6029312ull);
    bf16*  Wqb     = (bf16*)(ws + base + 6031360ull);
    bf16*  Wkb     = (bf16*)(ws + base + 6555648ull);
    bf16*  Wvb     = (bf16*)(ws + base + 7079936ull);
    float* out     = (float*)d_out;

    // embed+concat projection, LSTM
    x_kernel<<<256, 256, 0, stream>>>(y_t_1, c_t_1, emb_W, xc_W, xc_b, x);
    gates_kernel<<<2048, 256, 0, stream>>>(x, h0, W_ih, W_hh, b_ih, b_hh, gates);
    lstm_elem_kernel<<<512, 256, 0, stream>>>(gates, c0, h_new, c_new);
    // attention (WMMA + async-LDS path)
    convert_w_kernel<<<3072, 256, 0, stream>>>(Wq, Wk, Wv, Wqb, Wkb, Wvb);
    qkv_proj_kernel<<<dim3(ROWS / 64, DMODEL / 64, 3), 256, 16384, stream>>>(enc, Wqb, Wkb, Wvb, Qb, Kb, Vb);
    attn_kernel<<<dim3(B_SZ, NHEAD), 256, 115200, stream>>>(Qb, Kb, Vb, ctxmean);
    ct_kernel<<<1024, 256, 0, stream>>>(ctxmean, Wo, c_t);
    // output head
    o1_kernel<<<512, 256, 0, stream>>>(h_new, c_t, o1_W, o1_b, out1);
    pgen_kernel<<<2, 256, 0, stream>>>(c_t, h_new, c_new, x, pg_W, pg_b, p_gen);
    logits_kernel<<<dim3(B_SZ / 64, (VOCAB + 63) / 64), 256, 0, stream>>>(out1, o2_W, o2_b, logits);
    final_kernel<<<B_SZ, 256, 0, stream>>>(logits, p_gen, c_t, extvoc, out);
    aux_out_kernel<<<3330, 256, 0, stream>>>(h_new, c_new, c_t, p_gen, coverage, out);
}
